// SelfAttention_38044820308072
// MI455X (gfx1250) — matrix operand order repaired and checked by reading, not compiled, not run on hardware
//
#include <hip/hip_runtime.h>

#define B_  8
#define CH_ 256
#define N_  2048

typedef __attribute__((ext_vector_type(16))) __bf16 v16bf;
typedef __attribute__((ext_vector_type(8)))  float  v8f;
typedef __attribute__((ext_vector_type(4)))  unsigned int u32x4;
typedef __attribute__((ext_vector_type(4)))  int i32x4;
typedef __attribute__((ext_vector_type(8)))  int i32x8;

union Frag { v16bf v; unsigned int u[8]; };

__device__ __forceinline__ unsigned short f2bf(float f) {
  unsigned int u = __float_as_uint(f);
  unsigned int r = (u + 0x7FFFu + ((u >> 16) & 1u)) >> 16;
  return (unsigned short)r;
}
__device__ __forceinline__ unsigned int pk2(float lo, float hi) {
  return (unsigned int)f2bf(lo) | ((unsigned int)f2bf(hi) << 16);
}

// ---------------- pre-pass: convert + transpose into workspace (bf16) ------
// KT[b][n][c] = bf16(K[b][c][n])   (key-major rows, contiguous 512B/row)
__global__ __launch_bounds__(256)
void kt_prep(const float* __restrict__ K, unsigned short* __restrict__ KT) {
  const int n = blockIdx.x, b = blockIdx.y, c = threadIdx.x;
  KT[((size_t)b * N_ + n) * CH_ + c] = f2bf(K[((size_t)b * CH_ + c) * N_ + n]);
}
// VT[b][c][n] = bf16(V[b][n][c])   (ch-major rows, keys contiguous)
__global__ __launch_bounds__(256)
void vt_prep(const float* __restrict__ V, unsigned short* __restrict__ VT) {
  const int n = blockIdx.x * 256 + threadIdx.x, c = blockIdx.y, b = blockIdx.z;
  VT[((size_t)b * CH_ + c) * N_ + n] = f2bf(V[((size_t)b * N_ + n) * CH_ + c]);
}

// ---------------- TDM: 2D tile load (tile == tensor), LDS row padding -------
__device__ __forceinline__ void tdm_load_2d(unsigned lds_off, const unsigned short* gptr,
                                            unsigned dim0 /*elems per row*/,
                                            unsigned dim1 /*rows*/,
                                            unsigned stride0 /*elems*/,
                                            unsigned pad_i, unsigned pad_a) {
  const unsigned long long ga = (unsigned long long)gptr;
  u32x4 g0;
  g0[0] = 1u;                                   // count=1, user descriptor
  g0[1] = lds_off;                              // lds_addr
  g0[2] = (unsigned)(ga & 0xFFFFFFFFu);         // global_addr[31:0]
  g0[3] = (unsigned)((ga >> 32) & 0x01FFFFFFu) | (2u << 30);  // addr[56:32] | type=2
  i32x8 g1;
  g1[0] = (int)((1u << 16) | (1u << 20) | (pad_i << 22) | (pad_a << 25)); // data_size=2B, pad_en
  g1[1] = (int)(dim0 << 16);                    // tensor_dim0[15:0]
  g1[2] = (int)((dim0 >> 16) | (dim1 << 16));   // tensor_dim0[31:16] | tensor_dim1[15:0]
  g1[3] = (int)((dim1 >> 16) | (dim0 << 16));   // tensor_dim1[31:16] | tile_dim0
  g1[4] = (int)dim1;                            // tile_dim1 (tile_dim2 = 0)
  g1[5] = (int)stride0;                         // tensor_dim0_stride[31:0]
  g1[6] = 0; g1[7] = 0;
  const i32x4 z4 = {0, 0, 0, 0};
  const i32x8 z8 = {0, 0, 0, 0, 0, 0, 0, 0};
  __builtin_amdgcn_tensor_load_to_lds(g0, g1, z4, z4, z8, 0);
}

// ---------------- main flash-attention kernel ------------------------------
__global__ __launch_bounds__(128)
void attn_fa_kernel(const float* __restrict__ Q,
                    const unsigned short* __restrict__ KT,
                    const unsigned short* __restrict__ VT,
                    float* __restrict__ O) {
  constexpr int KT_S = 264;            // Kt row stride (elems): 512B data + 16B pad
  constexpr int VT_S = 40;             // Vt row stride (elems): 64B data + 16B pad
  constexpr int KT_TILE = 32 * KT_S;   // per-buffer Kt elems
  constexpr int VT_TILE = 256 * VT_S;  // per-buffer Vt elems
  __shared__ __align__(16) unsigned short Kt[2 * KT_TILE];
  __shared__ __align__(16) unsigned short Vt[2 * VT_TILE];
  __shared__ __align__(16) unsigned short Qs[64 * KT_S];
  __shared__ __align__(16) unsigned short Pl[4 * 16 * VT_S];

  const int tid  = threadIdx.x;
  const int wave = tid >> 5;
  const int lane = tid & 31;
  const int half = lane >> 4;
  const int l16  = lane & 15;
  const int b    = blockIdx.y;
  const int m0   = blockIdx.x * 64;            // first query row of this block

  // ---- stage Q (64 rows) once: global f32 -> LDS bf16, row-major
  {
    const float* qb = Q + ((size_t)b * N_ + m0) * CH_;
    for (int idx = tid; idx < 64 * 64; idx += 128) {
      const int row = idx >> 6, qd = idx & 63;
      float4 qv = *(const float4*)(qb + (size_t)row * CH_ + qd * 4);
      unsigned int* dst = (unsigned int*)&Qs[row * KT_S + qd * 4];
      dst[0] = pk2(qv.x, qv.y);
      dst[1] = pk2(qv.z, qv.w);
    }
  }

  const unsigned kt_lds = (unsigned)(unsigned long long)(void*)&Kt[0];
  const unsigned vt_lds = (unsigned)(unsigned long long)(void*)&Vt[0];
  const unsigned short* ktg = KT + (size_t)b * N_ * CH_;   // rows = keys
  const unsigned short* vtg = VT + (size_t)b * CH_ * N_;   // rows = channels

  // ---- preload first K/V tiles via Tensor Data Mover (wave 0 issues)
  if (wave == 0) {
    tdm_load_2d(kt_lds, ktg, 256u, 32u, 256u, 6u, 3u);            // K: 32 rows x 512B
    tdm_load_2d(vt_lds, vtg, 32u, 256u, (unsigned)N_, 3u, 3u);    // V: 256 rows x 64B
  }

  v8f acc[16];
  #pragma unroll
  for (int t = 0; t < 16; ++t) {
    v8f z = {0.f, 0.f, 0.f, 0.f, 0.f, 0.f, 0.f, 0.f};
    acc[t] = z;
  }
  float mrow[8], lrow[8];
  #pragma unroll
  for (int g = 0; g < 8; ++g) { mrow[g] = -1e30f; lrow[g] = 0.f; }

  const float scale = 0.0625f;  // 256^-0.5
  unsigned short* pw = &Pl[wave * 16 * VT_S];
  const unsigned short* qrow = &Qs[(wave * 16 + l16) * KT_S];

  for (int kb = 0; kb < N_; kb += 32) {
    const int cur = (kb >> 5) & 1;
    const int nxt = cur ^ 1;

    __syncthreads();  // all waves done reading buffer 'nxt' in the previous block
    if (wave == 0) {
      if (kb + 32 < N_) {
        tdm_load_2d(kt_lds + nxt * (KT_TILE * 2), ktg + (size_t)(kb + 32) * CH_,
                    256u, 32u, 256u, 6u, 3u);
        tdm_load_2d(vt_lds + nxt * (VT_TILE * 2), vtg + (kb + 32),
                    32u, 256u, (unsigned)N_, 3u, 3u);
        __builtin_amdgcn_s_wait_tensorcnt(2);   // current tile's 2 TDMs done
      } else {
        __builtin_amdgcn_s_wait_tensorcnt(0);
      }
    }
    __syncthreads();  // current buffers staged for everyone

    const unsigned short* ktt = &Kt[cur * KT_TILE];
    const unsigned short* vtt = &Vt[cur * VT_TILE];

    // ---- S = scale * Q K^T : two 16x16 tiles over this 32-key block
    v8f s[2];
    {
      v8f s0 = {0.f, 0.f, 0.f, 0.f, 0.f, 0.f, 0.f, 0.f};
      v8f s1 = s0;
      const unsigned short* k0 = &ktt[l16 * KT_S + (half ? 16 : 0)];         // key tile 0
      const unsigned short* k1 = &ktt[(16 + l16) * KT_S + (half ? 16 : 0)];  // key tile 1
      #pragma unroll
      for (int kk = 0; kk < 8; ++kk) {
        Frag af, b0, b1;
        {
          uint4 lo = *(const uint4*)(qrow + kk * 32 + (half ? 16 : 0));
          uint4 hi = *(const uint4*)(qrow + kk * 32 + (half ? 16 : 0) + 8);
          af.u[0] = lo.x; af.u[1] = lo.y; af.u[2] = lo.z; af.u[3] = lo.w;
          af.u[4] = hi.x; af.u[5] = hi.y; af.u[6] = hi.z; af.u[7] = hi.w;
        }
        {
          uint4 lo = *(const uint4*)(k0 + kk * 32);
          uint4 hi = *(const uint4*)(k0 + kk * 32 + 8);
          b0.u[0] = lo.x; b0.u[1] = lo.y; b0.u[2] = lo.z; b0.u[3] = lo.w;
          b0.u[4] = hi.x; b0.u[5] = hi.y; b0.u[6] = hi.z; b0.u[7] = hi.w;
        }
        {
          uint4 lo = *(const uint4*)(k1 + kk * 32);
          uint4 hi = *(const uint4*)(k1 + kk * 32 + 8);
          b1.u[0] = lo.x; b1.u[1] = lo.y; b1.u[2] = lo.z; b1.u[3] = lo.w;
          b1.u[4] = hi.x; b1.u[5] = hi.y; b1.u[6] = hi.z; b1.u[7] = hi.w;
        }
        s0 = __builtin_amdgcn_wmma_f32_16x16x32_bf16(false, af.v, false, b0.v,
                                                     (short)0, s0, false, false);
        s1 = __builtin_amdgcn_wmma_f32_16x16x32_bf16(false, af.v, false, b1.v,
                                                     (short)0, s1, false, false);
      }
      s[0] = s0; s[1] = s1;
    }

    // ---- online softmax (rows in VGPR index g; columns across 16-lane halves)
    float alpha[8];
    #pragma unroll
    for (int g = 0; g < 8; ++g) {
      float s0 = s[0][g] * scale, s1 = s[1][g] * scale;
      float t = fmaxf(s0, s1);
      t = fmaxf(t, __shfl_xor(t, 1, 16));
      t = fmaxf(t, __shfl_xor(t, 2, 16));
      t = fmaxf(t, __shfl_xor(t, 4, 16));
      t = fmaxf(t, __shfl_xor(t, 8, 16));
      const float mnew = fmaxf(mrow[g], t);
      alpha[g] = __expf(mrow[g] - mnew);
      mrow[g] = mnew;
      s0 = __expf(s0 - mnew);
      s1 = __expf(s1 - mnew);
      s[0][g] = s0; s[1][g] = s1;
      float r = s0 + s1;
      r += __shfl_xor(r, 1, 16);
      r += __shfl_xor(r, 2, 16);
      r += __shfl_xor(r, 4, 16);
      r += __shfl_xor(r, 8, 16);
      lrow[g] = lrow[g] * alpha[g] + r;
    }
    #pragma unroll
    for (int t = 0; t < 16; ++t) {
      #pragma unroll
      for (int g = 0; g < 8; ++g) acc[t][g] *= alpha[g];
    }

    // ---- P: C-layout f32 -> A-layout bf16 via per-wave LDS round-trip
    #pragma unroll
    for (int t = 0; t < 2; ++t) {
      #pragma unroll
      for (int g = 0; g < 8; ++g) {
        pw[(g + half * 8) * VT_S + t * 16 + l16] = f2bf(s[t][g]);
      }
    }
    __asm__ volatile("s_wait_dscnt 0" ::: "memory");

    Frag pf;
    {
      const unsigned short* pr = &pw[l16 * VT_S + (half ? 16 : 0)];
      uint4 lo = *(const uint4*)(pr);
      uint4 hi = *(const uint4*)(pr + 8);
      pf.u[0] = lo.x; pf.u[1] = lo.y; pf.u[2] = lo.z; pf.u[3] = lo.w;
      pf.u[4] = hi.x; pf.u[5] = hi.y; pf.u[6] = hi.z; pf.u[7] = hi.w;
    }

    // ---- acc += P V over 16 output-channel tiles
    #pragma unroll
    for (int t16 = 0; t16 < 16; ++t16) {
      Frag vf;
      const unsigned short* vr = &vtt[(t16 * 16 + l16) * VT_S + (half ? 16 : 0)];
      uint4 lo = *(const uint4*)(vr);
      uint4 hi = *(const uint4*)(vr + 8);
      vf.u[0] = lo.x; vf.u[1] = lo.y; vf.u[2] = lo.z; vf.u[3] = lo.w;
      vf.u[4] = hi.x; vf.u[5] = hi.y; vf.u[6] = hi.z; vf.u[7] = hi.w;
      acc[t16] = __builtin_amdgcn_wmma_f32_16x16x32_bf16(false, pf.v, false, vf.v,
                                                         (short)0, acc[t16], false, false);
    }
  }

  // ---- epilogue: divide by softmax sum, write transposed output [B][CH][N]
  float rl[8];
  #pragma unroll
  for (int g = 0; g < 8; ++g) rl[g] = 1.0f / lrow[g];

  float* obase = O + (size_t)b * CH_ * N_;
  const int mw = m0 + wave * 16;
  #pragma unroll
  for (int t16 = 0; t16 < 16; ++t16) {
    const int ch = t16 * 16 + l16;
    #pragma unroll
    for (int g = 0; g < 8; ++g) {
      obase[(size_t)ch * N_ + mw + g + half * 8] = acc[t16][g] * rl[g];
    }
  }
}

extern "C" void kernel_launch(void* const* d_in, const int* in_sizes, int n_in,
                              void* d_out, int out_size, void* d_ws, size_t ws_size,
                              hipStream_t stream) {
  const float* k = (const float*)d_in[0];
  const float* q = (const float*)d_in[1];
  const float* v = (const float*)d_in[2];
  float* o = (float*)d_out;

  unsigned short* ktg = (unsigned short*)d_ws;                    // 8 MiB
  unsigned short* vtg = ktg + (size_t)B_ * N_ * CH_;              // 8 MiB

  kt_prep<<<dim3(N_, B_, 1), dim3(256, 1, 1), 0, stream>>>(k, ktg);
  vt_prep<<<dim3(N_ / 256, CH_, B_), dim3(256, 1, 1), 0, stream>>>(v, vtg);
  attn_fa_kernel<<<dim3(N_ / 64, B_, 1), dim3(128, 1, 1), 0, stream>>>(q, ktg, vtg, o);
}
